// TraceModel_77996606095607
// MI455X (gfx1250) — matrix operand order repaired and checked by reading, not compiled, not run on hardware
//
#include <hip/hip_runtime.h>
#include <math.h>

#define T_SNAP 8
#define NNODE  10000
#define NEDGE  160000
#define FDIM   128
#define GGRAPH 32
#define RHID   128
#define NCLS   10

typedef __attribute__((ext_vector_type(16))) __bf16 v16bf;
typedef __attribute__((ext_vector_type(8)))  float  v8f;

union FragA { v16bf v; __bf16 e[16]; };

// -------------------------------------------------------------------------
// One-time weight prep: Wt[n][k] = (bf16) W[k][n]   (column-major bf16)
// Makes each lane's B fragment one contiguous 32B load in the GEMM.
// -------------------------------------------------------------------------
__global__ void w_to_bf16_t(const float* __restrict__ W, __bf16* __restrict__ Wt) {
  int i = blockIdx.x * blockDim.x + threadIdx.x;
  if (i >= FDIM * FDIM) return;
  int n = i >> 7, k = i & 127;
  Wt[i] = (__bf16)W[(size_t)k * FDIM + n];
}

// -------------------------------------------------------------------------
// Y[N,128] = X[N,128] @ W[128,128]  (bf16 WMMA, fp32 accumulate)
// One wave -> one 16x16 tile. Block = 8 waves = 16 rows x all 128 cols.
// A in contiguous-K order (same as B): lane holds row = lane%16; per K-chunk the
// lane's 16 values are two adjacent 8-element runs (K+khA, K+8+khA).
// B (from bf16 col-major Wt): lane holds col = lane%16; 16 contiguous K.
// -------------------------------------------------------------------------
__global__ __launch_bounds__(256) void gemm_128(const float* __restrict__ X,
                                                const __bf16* __restrict__ Wt,
                                                float* __restrict__ Y) {
  const int lane = threadIdx.x & 31;
  const int wave = threadIdx.x >> 5;
  const int m0 = blockIdx.x * 16;
  const int n0 = wave * 16;
  const int rowA = m0 + (lane & 15);
  const int colB = n0 + (lane & 15);
  const int khA  = (lane >> 4) << 4;   // lanes 16-31: K+16 for A
  const int khB  = (lane >> 4) << 4;   // lanes 16-31: K+16 for B
  const float*  arow = X + (size_t)rowA * FDIM;
  const __bf16* wcol = Wt + (size_t)colB * FDIM;
  v8f acc = {};
#pragma unroll
  for (int kc = 0; kc < 4; ++kc) {
    const int kbase = kc * 32;
    // ---- A fragment: 4x b128 fp32 loads, packed to bf16 ----
    const float* ap0 = arow + kbase + khA;        // K = kbase+khA .. +7
    const float* ap1 = ap0 + 8;                   // K = kbase+8+khA .. +7
    const float4 a00 = *(const float4*)(ap0);
    const float4 a01 = *(const float4*)(ap0 + 4);
    const float4 a10 = *(const float4*)(ap1);
    const float4 a11 = *(const float4*)(ap1 + 4);
    FragA a;
    a.e[0]  = (__bf16)a00.x; a.e[1]  = (__bf16)a00.y;
    a.e[2]  = (__bf16)a00.z; a.e[3]  = (__bf16)a00.w;
    a.e[4]  = (__bf16)a01.x; a.e[5]  = (__bf16)a01.y;
    a.e[6]  = (__bf16)a01.z; a.e[7]  = (__bf16)a01.w;
    a.e[8]  = (__bf16)a10.x; a.e[9]  = (__bf16)a10.y;
    a.e[10] = (__bf16)a10.z; a.e[11] = (__bf16)a10.w;
    a.e[12] = (__bf16)a11.x; a.e[13] = (__bf16)a11.y;
    a.e[14] = (__bf16)a11.z; a.e[15] = (__bf16)a11.w;
    // ---- B fragment: one contiguous 32B bf16 vector load ----
    const v16bf b = *(const v16bf*)(wcol + kbase + khB);
    acc = __builtin_amdgcn_wmma_f32_16x16x32_bf16(false, a.v, false, b,
                                                  (short)0, acc, false, false);
  }
#pragma unroll
  for (int r = 0; r < 8; ++r) {
    const int row = m0 + r + ((lane >> 4) << 3);  // VGPR r: M=r (lanes<16), M=r+8
    Y[(size_t)row * FDIM + n0 + (lane & 15)] = acc[r];
  }
}

// -------------------------------------------------------------------------
// small helpers
// -------------------------------------------------------------------------
__global__ void zero_f(float* p, int n) {
  int i = blockIdx.x * blockDim.x + threadIdx.x;
  if (i < n) p[i] = 0.f;
}
__global__ void fill1(float* p, int n) {
  int i = blockIdx.x * blockDim.x + threadIdx.x;
  if (i < n) p[i] = 1.f;
}
__global__ void deg_edges(const int* __restrict__ dst, float* __restrict__ deg) {
  int e = blockIdx.x * blockDim.x + threadIdx.x;
  if (e < NEDGE) atomicAdd(&deg[dst[e]], 1.f);
}
__global__ void rsqrt_k(float* p, int n) {
  int i = blockIdx.x * blockDim.x + threadIdx.x;
  if (i < n) p[i] = rsqrtf(p[i]);
}
// OUT[n,f] = H[n,f]*dinv[n]^2 + bias[f]   (self-loop term + bias)
__global__ void self_init(const float* __restrict__ H, const float* __restrict__ dinv,
                          const float* __restrict__ bias, float* __restrict__ OUT) {
  int i = blockIdx.x * blockDim.x + threadIdx.x;
  if (i >= NNODE * FDIM) return;
  int n = i >> 7, f = i & 127;
  float dv = dinv[n];
  OUT[i] = H[i] * dv * dv + bias[f];
}
// OUT[dst] += H[src] * dinv[src]*dinv[dst], 32 lanes/edge, float4 gather
__global__ void agg_edges(const float* __restrict__ H, const int* __restrict__ src,
                          const int* __restrict__ dst, const float* __restrict__ dinv,
                          float* __restrict__ OUT) {
  long long tid = (long long)blockIdx.x * blockDim.x + threadIdx.x;
  int e = (int)(tid >> 5);
  if (e >= NEDGE) return;
  int lane = (int)(tid & 31);
  int s = src[e], d = dst[e];
  float coef = dinv[s] * dinv[d];
  float4 hv = *(const float4*)(H + (size_t)s * FDIM + lane * 4);
  float* o = OUT + (size_t)d * FDIM + lane * 4;
  atomicAdd(o + 0, hv.x * coef);
  atomicAdd(o + 1, hv.y * coef);
  atomicAdd(o + 2, hv.z * coef);
  atomicAdd(o + 3, hv.w * coef);
}
__global__ void relu_k(float* p, int n) {
  int i = blockIdx.x * blockDim.x + threadIdx.x;
  if (i < n) p[i] = fmaxf(p[i], 0.f);
}
// graph mean-pool accumulation: sums[batch[n]] += H[n], cnt[batch[n]] += 1
__global__ void pool_sum(const float* __restrict__ H, const int* __restrict__ batch,
                         float* __restrict__ sums, float* __restrict__ cnt) {
  long long tid = (long long)blockIdx.x * blockDim.x + threadIdx.x;
  int n = (int)(tid >> 5);
  if (n >= NNODE) return;
  int lane = (int)(tid & 31);
  int g = batch[n];
  float4 hv = *(const float4*)(H + (size_t)n * FDIM + lane * 4);
  float* s = sums + (size_t)g * FDIM + lane * 4;
  atomicAdd(s + 0, hv.x);
  atomicAdd(s + 1, hv.y);
  atomicAdd(s + 2, hv.z);
  atomicAdd(s + 3, hv.w);
  if (lane == 0) atomicAdd(&cnt[g], 1.f);
}
__global__ void emb_fin(float* __restrict__ emb, const float* __restrict__ cnt) {
  int i = blockIdx.x * blockDim.x + threadIdx.x;
  if (i >= T_SNAP * GGRAPH * FDIM) return;
  emb[i] = emb[i] / fmaxf(cnt[i >> 7], 1.f);
}

// -------------------------------------------------------------------------
// One GRU step for all G graphs: block = one graph, thread = one hidden unit
// -------------------------------------------------------------------------
__global__ __launch_bounds__(RHID) void gru_step(const float* __restrict__ X,
                                                 float* __restrict__ Hst,
                                                 float* __restrict__ Outs,
                                                 const float* __restrict__ Wih,
                                                 const float* __restrict__ Whh,
                                                 const float* __restrict__ bih,
                                                 const float* __restrict__ bhh) {
  __shared__ float xg[RHID], hg[RHID];
  const int g = blockIdx.x, k = threadIdx.x;
  xg[k] = X[(size_t)g * RHID + k];
  hg[k] = Hst[(size_t)g * RHID + k];
  __syncthreads();
  float ir = bih[k], iz = bih[RHID + k], in_ = bih[2 * RHID + k];
  float hr = bhh[k], hz = bhh[RHID + k], hn = bhh[2 * RHID + k];
  const float* wr = Wih + (size_t)k * RHID;
  const float* wz = Wih + (size_t)(RHID + k) * RHID;
  const float* wn = Wih + (size_t)(2 * RHID + k) * RHID;
  const float* ur = Whh + (size_t)k * RHID;
  const float* uz = Whh + (size_t)(RHID + k) * RHID;
  const float* un = Whh + (size_t)(2 * RHID + k) * RHID;
  for (int j = 0; j < RHID; ++j) {
    float xj = xg[j], hj = hg[j];
    ir += xj * wr[j];  iz += xj * wz[j];  in_ += xj * wn[j];
    hr += hj * ur[j];  hz += hj * uz[j];  hn  += hj * un[j];
  }
  float r = 1.f / (1.f + expf(-(ir + hr)));
  float z = 1.f / (1.f + expf(-(iz + hz)));
  float n = tanhf(in_ + r * hn);
  float hnew = (1.f - z) * n + z * hg[k];
  Hst[(size_t)g * RHID + k] = hnew;
  if (Outs) Outs[(size_t)g * RHID + k] = hnew;
}

__global__ void classify(const float* __restrict__ H, const float* __restrict__ Wc,
                         const float* __restrict__ bc, float* __restrict__ out) {
  int i = blockIdx.x * blockDim.x + threadIdx.x;
  if (i >= GGRAPH * NCLS) return;
  int g = i / NCLS, c = i % NCLS;
  float acc = bc[c];
  for (int k = 0; k < RHID; ++k) acc += H[(size_t)g * RHID + k] * Wc[(size_t)c * RHID + k];
  out[i] = acc;
}

// -------------------------------------------------------------------------
extern "C" void kernel_launch(void* const* d_in, const int* in_sizes, int n_in,
                              void* d_out, int out_size, void* d_ws, size_t ws_size,
                              hipStream_t stream) {
  (void)in_sizes; (void)n_in; (void)out_size; (void)ws_size;
  const float* x    = (const float*)d_in[0];   // [T,N,128]
  const int*   ei   = (const int*)d_in[1];     // [T,2,E]
  const int*   bat  = (const int*)d_in[2];     // [T,N]
  const float* W1   = (const float*)d_in[3];
  const float* b1   = (const float*)d_in[4];
  const float* W2   = (const float*)d_in[5];
  const float* b2   = (const float*)d_in[6];
  const float* Wih0 = (const float*)d_in[7];
  const float* Whh0 = (const float*)d_in[8];
  const float* bih0 = (const float*)d_in[9];
  const float* bhh0 = (const float*)d_in[10];
  const float* Wih1 = (const float*)d_in[11];
  const float* Whh1 = (const float*)d_in[12];
  const float* bih1 = (const float*)d_in[13];
  const float* bhh1 = (const float*)d_in[14];
  const float* Wc   = (const float*)d_in[15];
  const float* bc   = (const float*)d_in[16];
  float* out = (float*)d_out;

  float* ws   = (float*)d_ws;
  float* A    = ws;                                 // [N,128] matmul output
  float* B    = A + (size_t)NNODE * FDIM;           // [N,128] aggregated features
  float* dinv = B + (size_t)NNODE * FDIM;           // [N]
  float* emb  = dinv + NNODE;                       // [T,G,128] pooled sums -> means
  float* cnt  = emb + (size_t)T_SNAP * GGRAPH * FDIM;  // [T,G]
  float* outs = cnt + (size_t)T_SNAP * GGRAPH;      // [T,G,128] GRU layer0 outputs
  float* h0   = outs + (size_t)T_SNAP * GGRAPH * FDIM; // [G,128]
  float* h1   = h0 + (size_t)GGRAPH * RHID;         // [G,128]
  __bf16* Wt1 = (__bf16*)(h1 + (size_t)GGRAPH * RHID); // [128,128] bf16 col-major
  __bf16* Wt2 = Wt1 + (size_t)FDIM * FDIM;

  // one-time (per call) weight conversion to bf16 column-major
  w_to_bf16_t<<<(FDIM * FDIM + 255) / 256, 256, 0, stream>>>(W1, Wt1);
  w_to_bf16_t<<<(FDIM * FDIM + 255) / 256, 256, 0, stream>>>(W2, Wt2);

  // zero everything from emb through h1 (includes outs; harmless, deterministic)
  const int zlen = T_SNAP * GGRAPH * FDIM + T_SNAP * GGRAPH +
                   T_SNAP * GGRAPH * FDIM + 2 * GGRAPH * RHID;
  zero_f<<<(zlen + 255) / 256, 256, 0, stream>>>(emb, zlen);

  const int NF = NNODE * FDIM;
  for (int t = 0; t < T_SNAP; ++t) {
    const float* xt  = x + (size_t)t * NNODE * FDIM;
    const int* srcE  = ei + (size_t)t * 2 * NEDGE;
    const int* dstE  = srcE + NEDGE;
    const int* bt    = bat + (size_t)t * NNODE;
    float* embT = emb + (size_t)t * GGRAPH * FDIM;
    float* cntT = cnt + (size_t)t * GGRAPH;

    // dinv = rsqrt(1 + in-degree)
    fill1<<<(NNODE + 255) / 256, 256, 0, stream>>>(dinv, NNODE);
    deg_edges<<<(NEDGE + 255) / 256, 256, 0, stream>>>(dstE, dinv);
    rsqrt_k<<<(NNODE + 255) / 256, 256, 0, stream>>>(dinv, NNODE);

    // layer 1: A = xt @ W1 ; B = A*dinv^2 + b1 ; B += scatter(A) ; relu
    gemm_128<<<NNODE / 16, 256, 0, stream>>>(xt, Wt1, A);
    self_init<<<(NF + 255) / 256, 256, 0, stream>>>(A, dinv, b1, B);
    agg_edges<<<(NEDGE * 32) / 256, 256, 0, stream>>>(A, srcE, dstE, dinv, B);
    relu_k<<<(NF + 255) / 256, 256, 0, stream>>>(B, NF);

    // layer 2
    gemm_128<<<NNODE / 16, 256, 0, stream>>>(B, Wt2, A);
    self_init<<<(NF + 255) / 256, 256, 0, stream>>>(A, dinv, b2, B);
    agg_edges<<<(NEDGE * 32) / 256, 256, 0, stream>>>(A, srcE, dstE, dinv, B);
    relu_k<<<(NF + 255) / 256, 256, 0, stream>>>(B, NF);

    // global mean pool (accumulate)
    pool_sum<<<(NNODE * 32 + 255) / 256, 256, 0, stream>>>(B, bt, embT, cntT);
  }
  emb_fin<<<(T_SNAP * GGRAPH * FDIM + 255) / 256, 256, 0, stream>>>(emb, cnt);

  // GRU layer 0 (writes outs), then layer 1 (final hidden state in h1)
  for (int t = 0; t < T_SNAP; ++t)
    gru_step<<<GGRAPH, RHID, 0, stream>>>(emb + (size_t)t * GGRAPH * FDIM, h0,
                                          outs + (size_t)t * GGRAPH * FDIM,
                                          Wih0, Whh0, bih0, bhh0);
  for (int t = 0; t < T_SNAP; ++t)
    gru_step<<<GGRAPH, RHID, 0, stream>>>(outs + (size_t)t * GGRAPH * FDIM, h1,
                                          nullptr, Wih1, Whh1, bih1, bhh1);

  classify<<<(GGRAPH * NCLS + 63) / 64, 64, 0, stream>>>(h1, Wc, bc, out);
}
